// ASAP_Pooling_29042568855970
// MI455X (gfx1250) — compile-verified, matrix operand order checked by reading, not run on hardware
//
#include <hip/hip_runtime.h>
#include <hip/hip_bf16.h>

typedef __attribute__((ext_vector_type(16))) _Float16 v16h;
typedef __attribute__((ext_vector_type(8)))  float    v8f;

#define TPB 256

// ---------- helpers ----------
__device__ __forceinline__ int f32_order(float f) {
    int i = __float_as_int(f);
    return (i >= 0) ? i : (i ^ 0x7FFFFFFF);
}
__device__ __forceinline__ float f32_unorder(int k) {
    return __int_as_float((k >= 0) ? k : (k ^ 0x7FFFFFFF));
}

// ---------- GEMM: D[N,64] = A[N,64] @ W[64,64] (+bias), f16 WMMA, f32 acc ----
// W is staged into LDS with CDNA5 async loads (ASYNCcnt), then pre-swizzled
// into the exact WMMA B-fragment VGPR layout so each fragment is one aligned
// 32B LDS read (2x ds_load_b128) per lane.
__global__ __launch_bounds__(128)
void gemm64_wmma(const float* A, const float* __restrict__ W,
                 const float* __restrict__ bias, float* D, int N)
{
    __shared__ float sWf[64 * 64];                       // 16KB staged f32 W
    __shared__ __align__(32) _Float16 sB[8][32][16];     // 8KB swizzled frags

    // --- async-stage W (16KB) into LDS: 1024 x b128 chunks, 8 per thread ---
    for (int i = threadIdx.x; i < 1024; i += 128) {
        const float* gsrc = W + i * 4;
        unsigned la = (unsigned)(size_t)(&sWf[i * 4]);   // flat[31:0] = LDS addr
        asm volatile("global_load_async_to_lds_b128 %0, %1, off"
                     :: "v"(la), "v"(gsrc) : "memory");
    }
    asm volatile("s_wait_asynccnt 0" ::: "memory");
    __syncthreads();

    // --- swizzle into B-fragment layout: frag = ct*2+kh, per-lane 16 halves --
    // lane l (0-15):  n = ct*16+l,     k = kh*32 + {0..7,16..23}
    // lane l (16-31): n = ct*16+(l-15? l&15), k = kh*32 + {8..15,24..31}
    for (int chunk = threadIdx.x; chunk < 512; chunk += 128) {
        int frag = chunk >> 6;            // 0..7
        int rem  = chunk & 63;
        int lane = rem >> 1;              // 0..31
        int part = rem & 1;               // j 0..7 vs 8..15
        int ct = frag >> 1, kh = frag & 1;
        int n = ct * 16 + (lane & 15);
        int kbase = kh * 32 + (part ? 16 : 0) + ((lane >= 16) ? 8 : 0);
        _Float16* dst = &sB[frag][lane][part * 8];
#pragma unroll
        for (int u = 0; u < 8; ++u)
            dst[u] = (_Float16)sWf[(kbase + u) * 64 + n];
    }
    __syncthreads();

    const int wave = threadIdx.x >> 5;
    const int lane = threadIdx.x & 31;
    const int laneHi = lane >> 4;
    const int l = lane & 15;
    const int rowBase = (blockIdx.x * 4 + wave) * 16;

    // --- A fragments (16x32 f16 layout, ISA 7.12.2): two runs of 8 floats ---
    v16h afrag[2];
    const int m = rowBase + l;
    const bool valid = (m < N);
    const float* arow = A + (size_t)m * 64;
    const float4 z4 = make_float4(0.f, 0.f, 0.f, 0.f);
#pragma unroll
    for (int kh = 0; kh < 2; ++kh) {
        int k0 = kh * 32 + (laneHi ? 8 : 0);
        float4 p0 = valid ? *(const float4*)(arow + k0)      : z4;
        float4 p1 = valid ? *(const float4*)(arow + k0 + 4)  : z4;
        float4 p2 = valid ? *(const float4*)(arow + k0 + 16) : z4;
        float4 p3 = valid ? *(const float4*)(arow + k0 + 20) : z4;
        float av[16] = { p0.x, p0.y, p0.z, p0.w, p1.x, p1.y, p1.z, p1.w,
                         p2.x, p2.y, p2.z, p2.w, p3.x, p3.y, p3.z, p3.w };
#pragma unroll
        for (int j = 0; j < 16; ++j)
            afrag[kh][j] = (_Float16)av[j];
    }

    const bool hasBias = (bias != nullptr);
#pragma unroll
    for (int ct = 0; ct < 4; ++ct) {
        v8f acc = {};
        const int n = ct * 16 + l;
#pragma unroll
        for (int kh = 0; kh < 2; ++kh) {
            v16h bfrag = *(const v16h*)&sB[ct * 2 + kh][lane][0];
            acc = __builtin_amdgcn_wmma_f32_16x16x32_f16(
                false, afrag[kh], false, bfrag, (short)0, acc, false, false);
        }
        float bv = hasBias ? bias[n] : 0.0f;
        float* dcol = D + (size_t)(rowBase + (laneHi ? 8 : 0)) * 64 + n;
        if (rowBase + 16 <= N) {          // full tile: unguarded stores
#pragma unroll
            for (int r = 0; r < 8; ++r)
                dcol[(size_t)r * 64] = acc[r] + bv;
        } else {
#pragma unroll
            for (int r = 0; r < 8; ++r) {
                int mm = rowBase + r + (laneHi ? 8 : 0);
                if (mm < N) D[(size_t)mm * 64 + n] = acc[r] + bv;
            }
        }
    }
}

// ---------- init / elementwise ----------
__global__ void init_nodes_k(float* deg, float* sexp, float* le, int* mkey, int N) {
    int i = blockIdx.x * blockDim.x + threadIdx.x;
    if (i >= N) return;
    deg[i] = 1.0f;            // self-loop weight
    sexp[i] = 0.0f;
    le[i] = 0.0f;
    mkey[i] = (int)0x80000000;
}
__global__ void init_nc_k(float* xpool, int* xqkey, int NC) {
    int i = blockIdx.x * blockDim.x + threadIdx.x;
    if (i >= NC) return;
    xpool[i] = 0.0f;
    xqkey[i] = (int)0x80000000;
}
__global__ void zero_nc_k(float* p, int NC) {
    int i = blockIdx.x * blockDim.x + threadIdx.x;
    if (i < NC) p[i] = 0.0f;
}
__global__ void deg_k(const int* row, const float* ew, float* deg, int E) {
    int e = blockIdx.x * blockDim.x + threadIdx.x;
    if (e < E) atomicAdd(&deg[row[e]], ew[e]);
}
__global__ void dinv_k(const float* deg, float* dinv, int N) {
    int i = blockIdx.x * blockDim.x + threadIdx.x;
    if (i < N) { float d = deg[i]; dinv[i] = (d > 0.0f) ? rsqrtf(d) : 0.0f; }
}
// x_pool += norm * h[col]  over all edges + self loops, per channel
__global__ void xpool_acc_k(const int* row, const int* col, const float* ew,
                            const float* dinv, const float* h, float* xpool,
                            int E, int N) {
    int t = blockIdx.x * blockDim.x + threadIdx.x;
    int EA = E + N;
    if (t >= EA * 64) return;
    int e = t >> 6, c = t & 63;
    int r, cl; float w;
    if (e < E) { r = row[e]; cl = col[e]; w = ew[e]; }
    else       { r = cl = e - E; w = 1.0f; }
    float norm = dinv[r] * w * dinv[cl];
    atomicAdd(&xpool[(size_t)r * 64 + c], norm * h[(size_t)cl * 64 + c]);
}
__global__ void add_bias_nc_k(float* xpool, const float* bg, int NC) {
    int i = blockIdx.x * blockDim.x + threadIdx.x;
    if (i < NC) xpool[i] += bg[i & 63];
}
// X_q = segment_max(x_pool[col]) per target row (ordered-int atomic max)
__global__ void xq_max_k(const int* row, const int* col, const float* xpool,
                         int* xqkey, int E, int N) {
    int t = blockIdx.x * blockDim.x + threadIdx.x;
    int EA = E + N;
    if (t >= EA * 64) return;
    int e = t >> 6, c = t & 63;
    int r, cl;
    if (e < E) { r = row[e]; cl = col[e]; } else { r = cl = e - E; }
    atomicMax(&xqkey[(size_t)r * 64 + c], f32_order(xpool[(size_t)cl * 64 + c]));
}
__global__ void unorder_k(int* buf, int n) {
    int i = blockIdx.x * blockDim.x + threadIdx.x;
    if (i < n) { float f = f32_unorder(buf[i]); buf[i] = __float_as_int(f); }
}
// per-node dots: q = <M_q, Wa[:64]>, p = <x_pool, Wa[64:]>
__global__ void node_qp_k(const float* mq, const float* xpool, const float* Wa,
                          float* q, float* p, int N) {
    int i = blockIdx.x * blockDim.x + threadIdx.x;
    if (i >= N) return;
    float sq = 0.0f, sp = 0.0f;
    for (int c = 0; c < 64; ++c) {
        sq += mq[(size_t)i * 64 + c] * Wa[c];
        sp += xpool[(size_t)i * 64 + c] * Wa[64 + c];
    }
    q[i] = sq; p[i] = sp;
}
// score = leaky_relu(q[row]+p[col]+ba); also segment max into mkey
__global__ void score_k(const int* row, const int* col, const float* q,
                        const float* p, const float* ba, float* score,
                        int* mkey, int E, int N) {
    int e = blockIdx.x * blockDim.x + threadIdx.x;
    int EA = E + N;
    if (e >= EA) return;
    int r, cl;
    if (e < E) { r = row[e]; cl = col[e]; } else { r = cl = e - E; }
    float s = q[r] + p[cl] + ba[0];
    s = (s > 0.0f) ? s : 0.2f * s;
    score[e] = s;
    atomicMax(&mkey[r], f32_order(s));
}
// e = exp(score - m[row]); score <- e; sexp[row] += e
__global__ void expsum_k(const int* row, float* score, const float* mf,
                         float* sexp, int E, int N) {
    int e = blockIdx.x * blockDim.x + threadIdx.x;
    int EA = E + N;
    if (e >= EA) return;
    int r = (e < E) ? row[e] : (e - E);
    float v = expf(score[e] - mf[r]);
    score[e] = v;
    atomicAdd(&sexp[r], v);
}
// out[row] += x[col] * (e / sexp[row])
__global__ void out_acc_k(const int* row, const int* col, const float* x,
                          const float* score, const float* sexp, float* out,
                          int E, int N) {
    int t = blockIdx.x * blockDim.x + threadIdx.x;
    int EA = E + N;
    if (t >= EA * 64) return;
    int e = t >> 6, c = t & 63;
    int r, cl;
    if (e < E) { r = row[e]; cl = col[e]; } else { r = cl = e - E; }
    float att = score[e] / sexp[r];
    atomicAdd(&out[(size_t)r * 64 + c], x[(size_t)cl * 64 + c] * att);
}
// per-node a1=<out,w1>, a2=<out,w2>, a3=<out,w3>
__global__ void node_a123_k(const float* out, const float* w1, const float* w2,
                            const float* w3, float* a1, float* a2, float* a3, int N) {
    int i = blockIdx.x * blockDim.x + threadIdx.x;
    if (i >= N) return;
    float s1 = 0.0f, s2 = 0.0f, s3 = 0.0f;
    for (int c = 0; c < 64; ++c) {
        float v = out[(size_t)i * 64 + c];
        s1 += v * w1[c]; s2 += v * w2[c]; s3 += v * w3[c];
    }
    a1[i] = s1; a2[i] = s2; a3[i] = s3;
}
__global__ void le_acc_k(const int* row, const int* col, const float* a1,
                         const float* a2, float* le, int E, int N) {
    int e = blockIdx.x * blockDim.x + threadIdx.x;
    int EA = E + N;
    if (e >= EA) return;
    int r, cl;
    if (e < E) { r = row[e]; cl = col[e]; } else { r = cl = e - E; }
    atomicAdd(&le[r], a1[cl] - a2[r]);
}
__global__ void fitness_k(const float* le, const float* a3, const float* bs,
                          float* fit, int N) {
    int i = blockIdx.x * blockDim.x + threadIdx.x;
    if (i >= N) return;
    float z = le[i] + a3[i] + bs[0];
    fit[i] = 1.0f / (1.0f + expf(-z));
}
// sort keys: fitness bits (positive, sigmoid>0) || inverted index (tie->low idx)
__global__ void build_keys_k(const float* fit, unsigned long long* keys,
                             int N, int SORT_N) {
    int i = blockIdx.x * blockDim.x + threadIdx.x;
    if (i >= SORT_N) return;
    if (i < N) {
        unsigned fb = __float_as_uint(fit[i]);
        keys[i] = ((unsigned long long)fb << 32) |
                  (unsigned long long)(0xFFFFFFFFu - (unsigned)i);
    } else {
        keys[i] = 0ULL;
    }
}
__global__ void bitonic_k(unsigned long long* keys, int j, int k, int SORT_N) {
    int i = blockIdx.x * blockDim.x + threadIdx.x;
    if (i >= SORT_N) return;
    int ixj = i ^ j;
    if (ixj > i) {
        unsigned long long a = keys[i], b = keys[ixj];
        bool desc = ((i & k) == 0);
        if (desc ? (a < b) : (a > b)) { keys[i] = b; keys[ixj] = a; }
    }
}
__global__ void write_out_k(const unsigned long long* keys, const float* out,
                            float* dout, int topk) {
    int t = blockIdx.x * blockDim.x + threadIdx.x;
    if (t >= topk * 64) return;
    int i = t >> 6, c = t & 63;
    unsigned long long kk = keys[i];
    unsigned idx = 0xFFFFFFFFu - (unsigned)(kk & 0xFFFFFFFFull);
    float fit = __uint_as_float((unsigned)(kk >> 32));
    dout[t] = out[(size_t)idx * 64 + c] * fit;
    if (c == 0) dout[(size_t)topk * 64 + i] = (float)idx;  // perm (promoted dtype)
}

// ---------- launcher ----------
static inline size_t alignup(size_t x) { return (x + 511) & ~(size_t)511; }

extern "C" void kernel_launch(void* const* d_in, const int* in_sizes, int n_in,
                              void* d_out, int out_size, void* d_ws, size_t ws_size,
                              hipStream_t stream) {
    const float* x  = (const float*)d_in[0];
    const int*   ei = (const int*)  d_in[1];
    const float* ew = (const float*)d_in[2];
    const float* Wg = (const float*)d_in[3];
    const float* bg = (const float*)d_in[4];
    const float* Wq = (const float*)d_in[5];
    const float* bq = (const float*)d_in[6];
    const float* Wa = (const float*)d_in[7];
    const float* ba = (const float*)d_in[8];
    const float* w1 = (const float*)d_in[9];
    const float* w2 = (const float*)d_in[10];
    const float* w3 = (const float*)d_in[11];
    const float* bs = (const float*)d_in[12];
    float* dout = (float*)d_out;

    const int N  = in_sizes[0] / 64;
    const int E  = in_sizes[2];
    const int EA = E + N;
    const int NC = N * 64;
    const int TOPK = (N + 1) / 2;            // ceil(0.5 * N)
    int SORT_N = 1; while (SORT_N < N) SORT_N <<= 1;

    const int* row = ei;
    const int* col = ei + E;

    // workspace carving (buffers reused as noted)
    char* ws = (char*)d_ws;
    float* hbuf  = (float*)ws;                ws += alignup((size_t)NC * 4);  // h, later 'out'
    float* xpool = (float*)ws;                ws += alignup((size_t)NC * 4);
    int*   xqkey = (int*)ws;                  ws += alignup((size_t)NC * 4);  // X_q keys -> X_q -> M_q
    float* score = (float*)ws;                ws += alignup((size_t)EA * 4);
    unsigned long long* keys = (unsigned long long*)ws;
                                              ws += alignup((size_t)SORT_N * 8);
    float* deg  = (float*)ws; ws += alignup((size_t)N * 4);
    float* dinv = (float*)ws; ws += alignup((size_t)N * 4);
    float* qv   = (float*)ws; ws += alignup((size_t)N * 4);
    float* pv   = (float*)ws; ws += alignup((size_t)N * 4);
    int*   mkey = (int*)ws;   ws += alignup((size_t)N * 4);                   // -> m (float)
    float* sexp = (float*)ws; ws += alignup((size_t)N * 4);
    float* le   = (float*)ws; ws += alignup((size_t)N * 4);
    float* a1   = (float*)ws; ws += alignup((size_t)N * 4);
    float* a2   = (float*)ws; ws += alignup((size_t)N * 4);
    float* a3   = (float*)ws; ws += alignup((size_t)N * 4);
    float* fit  = (float*)ws; ws += alignup((size_t)N * 4);

    auto g = [](int n) { return dim3((unsigned)((n + TPB - 1) / TPB)); };
    const dim3 B(TPB);

    // init
    init_nodes_k<<<g(N), B, 0, stream>>>(deg, sexp, le, mkey, N);
    init_nc_k<<<g(NC), B, 0, stream>>>(xpool, xqkey, NC);

    // h = x @ Wg  (WMMA)
    gemm64_wmma<<<dim3((N + 63) / 64), dim3(128), 0, stream>>>(x, Wg, nullptr, hbuf, N);

    // weighted degree + normalization
    deg_k<<<g(E), B, 0, stream>>>(row, ew, deg, E);
    dinv_k<<<g(N), B, 0, stream>>>(deg, dinv, N);

    // x_pool = segsum(norm * h[col]) + bg
    xpool_acc_k<<<g(EA * 64), B, 0, stream>>>(row, col, ew, dinv, hbuf, xpool, E, N);
    zero_nc_k<<<g(NC), B, 0, stream>>>(hbuf, NC);        // hbuf now 'out' accumulator
    add_bias_nc_k<<<g(NC), B, 0, stream>>>(xpool, bg, NC);

    // X_q = segment_max(x_pool[col]);  M_q = X_q @ Wq + bq (in place, WMMA)
    xq_max_k<<<g(EA * 64), B, 0, stream>>>(row, col, xpool, xqkey, E, N);
    unorder_k<<<g(NC), B, 0, stream>>>(xqkey, NC);
    gemm64_wmma<<<dim3((N + 63) / 64), dim3(128), 0, stream>>>(
        (const float*)xqkey, Wq, bq, (float*)xqkey, N);
    const float* mq = (const float*)xqkey;

    // attention scores + segment softmax
    node_qp_k<<<g(N), B, 0, stream>>>(mq, xpool, Wa, qv, pv, N);
    score_k<<<g(EA), B, 0, stream>>>(row, col, qv, pv, ba, score, mkey, E, N);
    unorder_k<<<g(N), B, 0, stream>>>(mkey, N);
    expsum_k<<<g(EA), B, 0, stream>>>(row, score, (const float*)mkey, sexp, E, N);

    // out = segsum(x[col] * att)
    out_acc_k<<<g(EA * 64), B, 0, stream>>>(row, col, x, score, sexp, hbuf, E, N);

    // LEConv fitness
    node_a123_k<<<g(N), B, 0, stream>>>(hbuf, w1, w2, w3, a1, a2, a3, N);
    le_acc_k<<<g(EA), B, 0, stream>>>(row, col, a1, a2, le, E, N);
    fitness_k<<<g(N), B, 0, stream>>>(le, a3, bs, fit, N);

    // top-k via bitonic sort on 64-bit keys (descending; ties -> lower index)
    build_keys_k<<<g(SORT_N), B, 0, stream>>>(fit, keys, N, SORT_N);
    for (int k = 2; k <= SORT_N; k <<= 1)
        for (int j = k >> 1; j > 0; j >>= 1)
            bitonic_k<<<g(SORT_N), B, 0, stream>>>(keys, j, k, SORT_N);

    // gather + scale, write perm
    write_out_k<<<g(TOPK * 64), B, 0, stream>>>(keys, hbuf, dout, TOPK);
}
